// DesignerNetwork_79972291052116
// MI455X (gfx1250) — matrix operand order repaired and checked by reading, not compiled, not run on hardware
//
#include <hip/hip_runtime.h>
#include <math.h>
#include <stdint.h>

// ---------- types ----------
typedef __bf16 bf16_t;
typedef __attribute__((ext_vector_type(16))) __bf16 v16bf;
typedef __attribute__((ext_vector_type(8)))  __bf16 v8bf;
typedef __attribute__((ext_vector_type(8)))  float  v8f;

// ---------- problem constants ----------
#define BATCH 512
#define NNODE 64
#define SPHI  60          // S_PHI == S_RHO
#define KH    64          // padded hidden width (K for GRU GEMMs)
#define NG    192         // gate-padded 3*64 output width
#define KC    96          // padded concat width (60 + 10 + 10 -> 96)
#define NR    64          // padded rho width
#define ROLE_OFF 2560     // out: instr_prob[512*5] | role_prob[512*7*64] | value[512]
#define VAL_OFF  231936

// ---------- device helpers ----------
__device__ __forceinline__ float sigm_(float x)  { return 1.f / (1.f + __expf(-x)); }
__device__ __forceinline__ float tanh_(float x)  { float e = __expf(-2.f * x); return (1.f - e) / (1.f + e); }

__device__ __forceinline__ v8f zero_v8f() {
  v8f c;
#pragma unroll
  for (int i = 0; i < 8; ++i) c[i] = 0.f;
  return c;
}

// Async global->LDS staging (CDNA5 path, tracked by ASYNCcnt). 16B per lane per op;
// `bytes` must be a multiple of 4096 (= 256 threads * 16B). GVS addressing:
// mem = SADDR(64b) + VADDR(32b offset); VDST holds the LDS byte address.
__device__ __forceinline__ void async_stage(const void* g, void* lds, int bytes) {
  unsigned base = (unsigned)(uintptr_t)lds;
  for (int off = (int)threadIdx.x * 16; off < bytes; off += 256 * 16) {
    asm volatile("global_load_async_to_lds_b128 %0, %1, %2 offset:0"
                 :: "v"(base + (unsigned)off), "v"((unsigned)off), "s"(g)
                 : "memory");
  }
}
__device__ __forceinline__ void wait_async0() {
  asm volatile("s_wait_asynccnt 0x0" ::: "memory");
}

// A fragment (16x32 bf16, M x K): lane m holds K {0..7,16..23} (lanes 0-15)
// or {8..15,24..31} (lanes 16-31), 2 elems per VGPR.
__device__ __forceinline__ v16bf load_frag_a(const bf16_t* base, int ld) {
  const int lane = threadIdx.x & 31;
  const int m    = lane & 15;
  const int koff = (lane >> 4) << 3;
  const bf16_t* p = base + m * ld + koff;
  v8bf lo = *(const v8bf*)(p);
  v8bf hi = *(const v8bf*)(p + 16);
  return __builtin_shufflevector(lo, hi, 0,1,2,3,4,5,6,7,8,9,10,11,12,13,14,15);
}

// B fragment (32x16 bf16, K x N) from row-major W[n][k] (n-major): lane n=L%16,
// K {0..7,16..23} (lanes 0-15) or {8..15,24..31} (lanes 16-31).
__device__ __forceinline__ v16bf load_frag_b(const bf16_t* base, int ld) {
  const int lane = threadIdx.x & 31;
  const int n    = lane & 15;
  const int k0   = (lane >> 4) << 3;
  const bf16_t* p = base + n * ld + k0;
  v8bf lo = *(const v8bf*)(p);
  v8bf hi = *(const v8bf*)(p + 16);
  return __builtin_shufflevector(lo, hi, 0,1,2,3,4,5,6,7,8,9,10,11,12,13,14,15);
}

__device__ __forceinline__ v8f wmma_bf16(v16bf a, v16bf b, v8f c) {
  return __builtin_amdgcn_wmma_f32_16x16x32_bf16(false, a, false, b, (short)0, c, false, false);
}

__device__ __forceinline__ float wred_max(float v) {
#pragma unroll
  for (int o = 16; o > 0; o >>= 1) v = fmaxf(v, __shfl_xor(v, o));
  return v;
}
__device__ __forceinline__ float wred_sum(float v) {
#pragma unroll
  for (int o = 16; o > 0; o >>= 1) v += __shfl_xor(v, o);
  return v;
}

// ---------- weight prep: f32 -> padded bf16 ----------
// Wih/Whh (180x60) -> gate-padded [3*64][64]; Wf/Wb (60x80) -> [64][96]
__global__ void prep_weights(const float* __restrict__ WihF, const float* __restrict__ WhhF,
                             const float* __restrict__ WihB, const float* __restrict__ WhhB,
                             const float* __restrict__ Wf,   const float* __restrict__ Wb,
                             bf16_t* __restrict__ wihF, bf16_t* __restrict__ whhF,
                             bf16_t* __restrict__ wihB, bf16_t* __restrict__ whhB,
                             bf16_t* __restrict__ wfP,  bf16_t* __restrict__ wbP) {
  int t = blockIdx.x * blockDim.x + threadIdx.x;
  const int NA = NG * KH;        // 12288
  if (t < 4 * NA) {
    int a = t / NA, r = t % NA;
    int np = r / KH, k = r % KH;       // np: gate-padded output index
    int g = np >> 6, cc = np & 63;
    const float* src = (a == 0) ? WihF : (a == 1) ? WhhF : (a == 2) ? WihB : WhhB;
    bf16_t* dst = (a == 0) ? wihF : (a == 1) ? whhF : (a == 2) ? wihB : whhB;
    float v = (cc < SPHI && k < SPHI) ? src[(g * SPHI + cc) * SPHI + k] : 0.f;
    dst[np * KH + k] = (bf16_t)v;
  } else {
    int t2 = t - 4 * NA;
    const int NB = NR * KC;            // 6144
    if (t2 < 2 * NB) {
      int a = t2 / NB, r = t2 % NB;
      int n = r / KC, k = r % KC;
      const float* src = (a == 0) ? Wf : Wb;
      bf16_t* dst = (a == 0) ? wfP : wbP;
      float v = (n < SPHI && k < 80) ? src[n * 80 + k] : 0.f;
      dst[n * KC + k] = (bf16_t)v;
    }
  }
}

__global__ void zero_state(float* __restrict__ h32, bf16_t* __restrict__ h16, int n) {
  int t = blockIdx.x * blockDim.x + threadIdx.x;
  if (t < n) { h32[t] = 0.f; h16[t] = (bf16_t)0.f; }
}

// ---------- per-node kernel: rho[j] = act(concat(h_j, z_j, dz_j) @ Wf^T + b),
//            gi[j]  = rho[j] @ Wih^T + bih  (gate-padded [512][192]) ----------
__global__ __launch_bounds__(256)
void node_kernel(int j, int dir,
                 const float* __restrict__ h32,   // [512][64] slice of node j
                 const float* __restrict__ z, const float* __restrict__ dz,
                 const bf16_t* __restrict__ wP,   // [64][96]
                 const float* __restrict__ bvec,  // [60]
                 const bf16_t* __restrict__ wih,  // [192][64] gate-padded
                 const float* __restrict__ bih,   // [180]
                 float* __restrict__ rho_out,     // [512][60] slice of node j
                 bf16_t* __restrict__ rho16,      // [512][64]
                 float* __restrict__ gi,          // [512][192]
                 float* __restrict__ gi_tail) {   // nullable extra copy
  __shared__ bf16_t lds_a [128 * KC];   // concat tile, later reused as rho16 [128][64]
  __shared__ bf16_t lds_wf[NR * KC];    // Wf   (64x96)  12KB
  __shared__ bf16_t lds_wi[NG * KH];    // Wih (192x64)  24KB
  const int tid = threadIdx.x;
  const int rb  = blockIdx.x * 128;

  // kick off async weight staging (overlapped with concat-tile build below)
  async_stage(wP,  lds_wf, NR * KC * 2);
  async_stage(wih, lds_wi, NG * KH * 2);

  for (int e = tid; e < 128 * KC; e += 256) {
    int m = e / KC, c = e % KC;
    int row = rb + m;
    float v;
    if      (c < 60) v = h32[row * KH + c];
    else if (c < 70) v = z [row * 640 + j * 10 + (c - 60)];
    else if (c < 80) v = dz[row * 640 + j * 10 + (c - 70)];
    else             v = 0.f;
    lds_a[e] = (bf16_t)v;
  }
  wait_async0();
  __syncthreads();

  const int wave = tid >> 5, lane = tid & 31;
  const int hi8  = (lane >> 4) << 3;
  const int nidx = lane & 15;
  const int mt   = wave;

  v16bf a0 = load_frag_a(lds_a + (mt * 16) * KC +  0, KC);
  v16bf a1 = load_frag_a(lds_a + (mt * 16) * KC + 32, KC);
  v16bf a2 = load_frag_a(lds_a + (mt * 16) * KC + 64, KC);
  v8f acc[4];
#pragma unroll
  for (int nt = 0; nt < 4; ++nt) {
    v8f c = zero_v8f();
    c = wmma_bf16(a0, load_frag_b(lds_wf + (nt * 16) * KC +  0, KC), c);
    c = wmma_bf16(a1, load_frag_b(lds_wf + (nt * 16) * KC + 32, KC), c);
    c = wmma_bf16(a2, load_frag_b(lds_wf + (nt * 16) * KC + 64, KC), c);
    acc[nt] = c;
  }
  __syncthreads();   // all reads of lds_a (concat) done

  // rho epilogue: tanh for fwd, identity for bwd; stash bf16 tile for phase D
#pragma unroll
  for (int nt = 0; nt < 4; ++nt) {
#pragma unroll
    for (int v = 0; v < 8; ++v) {
      int m = mt * 16 + v + hi8;
      int c = nt * 16 + nidx;
      int row = rb + m;
      float val = 0.f;
      if (c < 60) {
        val = acc[nt][v] + bvec[c];
        if (dir == 0) val = tanh_(val);
        rho_out[row * SPHI + c] = val;
      }
      bf16_t bv = (bf16_t)val;
      rho16[row * NR + c] = bv;
      lds_a[m * NR + c]   = bv;
    }
  }
  __syncthreads();

  // gi = rho @ Wih^T + bih
  v16bf r0 = load_frag_a(lds_a + (mt * 16) * NR +  0, NR);
  v16bf r1 = load_frag_a(lds_a + (mt * 16) * NR + 32, NR);
#pragma unroll
  for (int nt = 0; nt < 12; ++nt) {
    v8f c = zero_v8f();
    c = wmma_bf16(r0, load_frag_b(lds_wi + (nt * 16) * KH +  0, KH), c);
    c = wmma_bf16(r1, load_frag_b(lds_wi + (nt * 16) * KH + 32, KH), c);
#pragma unroll
    for (int v = 0; v < 8; ++v) {
      int m  = mt * 16 + v + hi8;
      int np = nt * 16 + nidx;               // gate-padded column
      int g  = np >> 6, cc = np & 63;
      float bias = (cc < 60) ? bih[g * SPHI + cc] : 0.f;
      float val = c[v] + bias;
      int row = rb + m;
      gi[row * NG + np] = val;
      if (gi_tail) gi_tail[row * NG + np] = val;
    }
  }
}

// ---------- batched GRU step: for every live target i, h_i <- gated GRU ----------
__global__ __launch_bounds__(256)
void step_kernel(int j, int dir,
                 const int* __restrict__ adj,
                 float* __restrict__ h32all, bf16_t* __restrict__ h16all, // [64][512][64]
                 const bf16_t* __restrict__ whh,  // [192][64] gate-padded
                 const float* __restrict__ bhh,   // [180]
                 const float* __restrict__ gi,    // [512][192]
                 const float* __restrict__ has) { // [512][64]
  const int i = blockIdx.y;
  if (dir == 0) { if (i <= j || adj[j * NNODE + i] == 0) return; }
  else          { if (i >= j || adj[i * NNODE + j] == 0) return; }

  __shared__ bf16_t lds_w[NG * KH];
  async_stage(whh, lds_w, NG * KH * 2);

  const int tid = threadIdx.x, wave = tid >> 5, lane = tid & 31;
  const int hi8 = (lane >> 4) << 3, nidx = lane & 15, mt = wave;
  const int rb  = blockIdx.x * 128;
  float*  h32 = h32all + (size_t)i * BATCH * KH;
  bf16_t* h16 = h16all + (size_t)i * BATCH * KH;

  // overlap A-fragment global loads with the async weight copy
  v16bf a0 = load_frag_a(h16 + (size_t)(rb + mt * 16) * KH +  0, KH);
  v16bf a1 = load_frag_a(h16 + (size_t)(rb + mt * 16) * KH + 32, KH);

  wait_async0();
  __syncthreads();

#pragma unroll
  for (int ntg = 0; ntg < 4; ++ntg) {
    v8f aR = zero_v8f(), aZ = zero_v8f(), aN = zero_v8f();
    aR = wmma_bf16(a0, load_frag_b(lds_w + (0 * 64 + ntg * 16) * KH +  0, KH), aR);
    aR = wmma_bf16(a1, load_frag_b(lds_w + (0 * 64 + ntg * 16) * KH + 32, KH), aR);
    aZ = wmma_bf16(a0, load_frag_b(lds_w + (1 * 64 + ntg * 16) * KH +  0, KH), aZ);
    aZ = wmma_bf16(a1, load_frag_b(lds_w + (1 * 64 + ntg * 16) * KH + 32, KH), aZ);
    aN = wmma_bf16(a0, load_frag_b(lds_w + (2 * 64 + ntg * 16) * KH +  0, KH), aN);
    aN = wmma_bf16(a1, load_frag_b(lds_w + (2 * 64 + ntg * 16) * KH + 32, KH), aN);

    const int c = ntg * 16 + nidx;
    if (c < 60) {
      const float bhr = bhh[c], bhz = bhh[60 + c], bhn = bhh[120 + c];
#pragma unroll
      for (int v = 0; v < 8; ++v) {
        int row = rb + mt * 16 + v + hi8;
        const float* gr = gi + (size_t)row * NG;
        float r  = sigm_(gr[c]       + aR[v] + bhr);
        float zg = sigm_(gr[64 + c]  + aZ[v] + bhz);
        float nn = tanh_(gr[128 + c] + r * (aN[v] + bhn));
        float h  = h32[row * KH + c];
        float hn = (1.f - zg) * nn + zg * h;
        float bl = has[row * NNODE + j];
        float hw = h + bl * (hn - h);
        h32[row * KH + c] = hw;
        h16[row * KH + c] = (bf16_t)hw;
      }
    }
  }
}

// ---------- alpha chains (8 sequential gated GRU steps, batch-parallel) ----------
__global__ __launch_bounds__(256)
void alpha_kernel(int dir,
                  const float* __restrict__ gi_tail,  // [8][512][192]
                  const bf16_t* __restrict__ whh, const float* __restrict__ bhh,
                  const float* __restrict__ has,
                  float* __restrict__ alpha) {        // [512][60]
  __shared__ bf16_t lds_w[NG * KH];
  __shared__ bf16_t lds_h[128 * KH];
  const int tid = threadIdx.x;
  async_stage(whh, lds_w, NG * KH * 2);
  for (int e = tid; e < 128 * KH; e += 256) lds_h[e] = (bf16_t)0.f;
  wait_async0();
  __syncthreads();

  const int wave = tid >> 5, lane = tid & 31;
  const int hi8 = (lane >> 4) << 3, nidx = lane & 15, mt = wave;
  const int rb  = blockIdx.x * 128;
  float hreg[4][8];
#pragma unroll
  for (int a = 0; a < 4; ++a)
#pragma unroll
    for (int v = 0; v < 8; ++v) hreg[a][v] = 0.f;

  for (int s = 0; s < 8; ++s) {
    const int node = (dir == 0) ? (56 + s) : (7 - s);
    const int slot = (dir == 0) ? s : (7 - s);
    const float* gis = gi_tail + (size_t)slot * BATCH * NG;

    v16bf a0 = load_frag_a(lds_h + (mt * 16) * KH +  0, KH);
    v16bf a1 = load_frag_a(lds_h + (mt * 16) * KH + 32, KH);
#pragma unroll
    for (int ntg = 0; ntg < 4; ++ntg) {
      v8f aR = zero_v8f(), aZ = zero_v8f(), aN = zero_v8f();
      aR = wmma_bf16(a0, load_frag_b(lds_w + (0 * 64 + ntg * 16) * KH +  0, KH), aR);
      aR = wmma_bf16(a1, load_frag_b(lds_w + (0 * 64 + ntg * 16) * KH + 32, KH), aR);
      aZ = wmma_bf16(a0, load_frag_b(lds_w + (1 * 64 + ntg * 16) * KH +  0, KH), aZ);
      aZ = wmma_bf16(a1, load_frag_b(lds_w + (1 * 64 + ntg * 16) * KH + 32, KH), aZ);
      aN = wmma_bf16(a0, load_frag_b(lds_w + (2 * 64 + ntg * 16) * KH +  0, KH), aN);
      aN = wmma_bf16(a1, load_frag_b(lds_w + (2 * 64 + ntg * 16) * KH + 32, KH), aN);

      const int c = ntg * 16 + nidx;
      if (c < 60) {
        const float bhr = bhh[c], bhz = bhh[60 + c], bhn = bhh[120 + c];
#pragma unroll
        for (int v = 0; v < 8; ++v) {
          int m = mt * 16 + v + hi8;
          int row = rb + m;
          const float* gr = gis + (size_t)row * NG;
          float r  = sigm_(gr[c]       + aR[v] + bhr);
          float zg = sigm_(gr[64 + c]  + aZ[v] + bhz);
          float nn = tanh_(gr[128 + c] + r * (aN[v] + bhn));
          float h  = hreg[ntg][v];
          float hn = (1.f - zg) * nn + zg * h;
          float bl = has[row * NNODE + node];
          float hw = h + bl * (hn - h);
          hreg[ntg][v] = hw;
          lds_h[m * KH + c] = (bf16_t)hw;
        }
      }
    }
    __syncthreads();
  }

#pragma unroll
  for (int ntg = 0; ntg < 4; ++ntg) {
    const int c = ntg * 16 + nidx;
    if (c < 60) {
#pragma unroll
      for (int v = 0; v < 8; ++v)
        alpha[(size_t)(rb + mt * 16 + v + hi8) * SPHI + c] = hreg[ntg][v];
    }
  }
}

// ---------- heads: omega softmax + value ----------
__global__ void heads_kernel(const float* __restrict__ af, const float* __restrict__ ab,
                             const float* __restrict__ Wa, const float* __restrict__ ba,
                             const float* __restrict__ Wc, const float* __restrict__ bc,
                             float* __restrict__ out) {
  int row = blockIdx.x * blockDim.x + threadIdx.x;
  if (row >= BATCH) return;
  float acc[6];
#pragma unroll
  for (int a = 0; a < 5; ++a) acc[a] = ba[a];
  acc[5] = bc[0];
  for (int k = 0; k < 120; ++k) {
    float x = (k < 60) ? af[row * SPHI + k] : ab[row * SPHI + (k - 60)];
#pragma unroll
    for (int a = 0; a < 5; ++a) acc[a] += Wa[a * 120 + k] * x;
    acc[5] += Wc[k] * x;
  }
  float m = acc[0];
#pragma unroll
  for (int a = 1; a < 5; ++a) m = fmaxf(m, acc[a]);
  float e[5], s = 0.f;
#pragma unroll
  for (int a = 0; a < 5; ++a) { e[a] = __expf(acc[a] - m); s += e[a]; }
#pragma unroll
  for (int a = 0; a < 5; ++a) out[row * 5 + a] = e[a] / s;
  out[VAL_OFF + row] = acc[5];
}

// ---------- psi / role softmax: one wave per (b, r), lanes over nodes ----------
__global__ __launch_bounds__(256)
void psi_kernel(const float* __restrict__ rho_f, const float* __restrict__ rho_b,
                const float* __restrict__ Wu, const float* __restrict__ bu,
                const float* __restrict__ has, float* __restrict__ out) {
  int wid  = (blockIdx.x * blockDim.x + threadIdx.x) >> 5;
  int lane = threadIdx.x & 31;
  int b = wid / 7, r = wid % 7;
  float vals[2];
#pragma unroll
  for (int t = 0; t < 2; ++t) {
    int n = lane + t * 32;
    float d = bu[r];
    const float* rf = rho_f + (size_t)n * BATCH * SPHI + (size_t)b * SPHI;
    const float* rb = rho_b + (size_t)n * BATCH * SPHI + (size_t)b * SPHI;
    for (int k = 0; k < 60; ++k) d += Wu[r * 120 + k] * rf[k];
    for (int k = 0; k < 60; ++k) d += Wu[r * 120 + 60 + k] * rb[k];
    float m = has[b * NNODE + n];
    vals[t] = -60.f * (1.f - m) + d * m;
  }
  float mx = wred_max(fmaxf(vals[0], vals[1]));
  float e0 = __expf(vals[0] - mx), e1 = __expf(vals[1] - mx);
  float s = wred_sum(e0 + e1);
  float* dst = out + ROLE_OFF + (size_t)b * 7 * NNODE + (size_t)r * NNODE;
  dst[lane]      = e0 / s;
  dst[lane + 32] = e1 / s;
}

// ---------- host launch ----------
extern "C" void kernel_launch(void* const* d_in, const int* in_sizes, int n_in,
                              void* d_out, int out_size, void* d_ws, size_t ws_size,
                              hipStream_t stream) {
  (void)in_sizes; (void)n_in; (void)out_size; (void)ws_size;
  const float* z    = (const float*)d_in[0];
  const float* dz   = (const float*)d_in[1];
  const float* has  = (const float*)d_in[2];
  const int*   adj  = (const int*)  d_in[3];
  const float* WihF = (const float*)d_in[4];
  const float* WhhF = (const float*)d_in[5];
  const float* bihF = (const float*)d_in[6];
  const float* bhhF = (const float*)d_in[7];
  const float* WihB = (const float*)d_in[8];
  const float* WhhB = (const float*)d_in[9];
  const float* bihB = (const float*)d_in[10];
  const float* bhhB = (const float*)d_in[11];
  const float* Wf   = (const float*)d_in[12];
  const float* bfv  = (const float*)d_in[13];
  const float* Wb   = (const float*)d_in[14];
  const float* bbv  = (const float*)d_in[15];
  const float* Wa   = (const float*)d_in[16];
  const float* ba   = (const float*)d_in[17];
  const float* Wc   = (const float*)d_in[18];
  const float* bc   = (const float*)d_in[19];
  const float* Wu   = (const float*)d_in[20];
  const float* bu   = (const float*)d_in[21];
  float* out = (float*)d_out;

  char* ws = (char*)d_ws;
  size_t off = 0;
  auto alloc = [&](size_t bytes) { size_t o = off; off += (bytes + 255) & ~(size_t)255; return o; };
  float*  rhoF  = (float*) (ws + alloc((size_t)NNODE * BATCH * SPHI * 4));
  float*  rhoB  = (float*) (ws + alloc((size_t)NNODE * BATCH * SPHI * 4));
  float*  h32   = (float*) (ws + alloc((size_t)NNODE * BATCH * KH * 4));
  bf16_t* h16   = (bf16_t*)(ws + alloc((size_t)NNODE * BATCH * KH * 2));
  bf16_t* rho16 = (bf16_t*)(ws + alloc((size_t)BATCH * NR * 2));
  float*  gi    = (float*) (ws + alloc((size_t)BATCH * NG * 4));
  float*  giF   = (float*) (ws + alloc((size_t)8 * BATCH * NG * 4));
  float*  giB   = (float*) (ws + alloc((size_t)8 * BATCH * NG * 4));
  float*  aF    = (float*) (ws + alloc((size_t)BATCH * SPHI * 4));
  float*  aB    = (float*) (ws + alloc((size_t)BATCH * SPHI * 4));
  bf16_t* wihFP = (bf16_t*)(ws + alloc((size_t)NG * KH * 2));
  bf16_t* whhFP = (bf16_t*)(ws + alloc((size_t)NG * KH * 2));
  bf16_t* wihBP = (bf16_t*)(ws + alloc((size_t)NG * KH * 2));
  bf16_t* whhBP = (bf16_t*)(ws + alloc((size_t)NG * KH * 2));
  bf16_t* wfP   = (bf16_t*)(ws + alloc((size_t)NR * KC * 2));
  bf16_t* wbP   = (bf16_t*)(ws + alloc((size_t)NR * KC * 2));

  prep_weights<<<240, 256, 0, stream>>>(WihF, WhhF, WihB, WhhB, Wf, Wb,
                                        wihFP, whhFP, wihBP, whhBP, wfP, wbP);
  zero_state<<<8192, 256, 0, stream>>>(h32, h16, NNODE * BATCH * KH);

  for (int j = 0; j < NNODE; ++j) {                     // forward sweep
    node_kernel<<<4, 256, 0, stream>>>(j, 0, h32 + (size_t)j * BATCH * KH, z, dz,
        wfP, bfv, wihFP, bihF, rhoF + (size_t)j * BATCH * SPHI, rho16, gi,
        (j >= NNODE - 8) ? giF + (size_t)(j - (NNODE - 8)) * BATCH * NG : nullptr);
    if (j < NNODE - 1)
      step_kernel<<<dim3(4, NNODE), 256, 0, stream>>>(j, 0, adj, h32, h16, whhFP, bhhF, gi, has);
  }

  zero_state<<<8192, 256, 0, stream>>>(h32, h16, NNODE * BATCH * KH);

  for (int j = NNODE - 1; j >= 0; --j) {                // backward sweep
    node_kernel<<<4, 256, 0, stream>>>(j, 1, h32 + (size_t)j * BATCH * KH, z, dz,
        wbP, bbv, wihBP, bihB, rhoB + (size_t)j * BATCH * SPHI, rho16, gi,
        (j < 8) ? giB + (size_t)j * BATCH * NG : nullptr);
    if (j > 0)
      step_kernel<<<dim3(4, NNODE), 256, 0, stream>>>(j, 1, adj, h32, h16, whhBP, bhhB, gi, has);
  }

  alpha_kernel<<<4, 256, 0, stream>>>(0, giF, whhFP, bhhF, has, aF);
  alpha_kernel<<<4, 256, 0, stream>>>(1, giB, whhBP, bhhB, has, aB);
  heads_kernel<<<2, 256, 0, stream>>>(aF, aB, Wa, ba, Wc, bc, out);
  psi_kernel<<<448, 256, 0, stream>>>(rhoF, rhoB, Wu, bu, has, out);
}
